// InterGraphAttention_4501125726637
// MI455X (gfx1250) — matrix operand order repaired and checked by reading, not compiled, not run on hardware
//
#include <hip/hip_runtime.h>
#include <hip/hip_bf16.h>

// ---------------------------------------------------------------------------
// Bipartite GAT (InterGraphAttention) for MI455X / gfx1250 (wave32, WMMA).
//   N = 50000 nodes/side, E = 800000 edges, D = 256, H = 2, C = 32 (H*C = 64)
// Pipeline:
//   1) Bpack = f16([W_src | W_dst]) pre-swizzled into per-lane WMMA B
//      fragments (32 contiguous bytes per lane per k-step -> b128 loads)
//   2) xsxd_{h,t} = elu(X) @ W via 8x unrolled v_wmma_f32_16x16x32_f16,
//      two accumulator chains, A row-block staged once in LDS (8 KB)
//   3) a_s/a_d per (node, head) small dots
//   4) per direction: segment-max (ordered-uint atomicMax), segment-sum of
//      exp, then scatter-add alpha * xs[src] into d_out (f32 atomics -> L2)
//   5) mean(enrichment_weights) reduction, epilogue bias add
// ---------------------------------------------------------------------------

typedef __attribute__((ext_vector_type(16))) _Float16 v16h;
typedef __attribute__((ext_vector_type(8)))  _Float16 v8h;
typedef __attribute__((ext_vector_type(2)))  _Float16 v2h;
typedef __attribute__((ext_vector_type(8)))  float    v8f;

#define NEG_SLOPE 0.2f

__device__ __forceinline__ float elu_f(float x)   { return x > 0.f ? x : (expf(x) - 1.f); }
__device__ __forceinline__ float lrelu_f(float x) { return x > 0.f ? x : NEG_SLOPE * x; }

// order-preserving float <-> uint encode (for float atomicMax via uint atomicMax)
__device__ __forceinline__ unsigned f2ord(float f) {
  unsigned u = __float_as_uint(f);
  return (u >> 31) ? ~u : (u | 0x80000000u);
}
__device__ __forceinline__ float ord2f(unsigned u) {
  return __uint_as_float((u >> 31) ? (u ^ 0x80000000u) : ~u);
}

// -------------------------------------------------------------------- fills
__global__ void fill_f32_kernel(float* __restrict__ p, float v, int n) {
  int i = blockIdx.x * blockDim.x + threadIdx.x;
  if (i < n) p[i] = v;
}
__global__ void fill_ord_kernel(unsigned* __restrict__ p, float v, int n) {
  int i = blockIdx.x * blockDim.x + threadIdx.x;
  if (i < n) p[i] = f2ord(v);
}

// ---------------------- pack [W_src | W_dst] into WMMA-B fragment order, f16
// Packed index: (((kk*8 + ct)*32 + lane)*16 + t)
//   kk   = K block of 32 (0..7)
//   ct   = 16-wide column tile (0..7)
//   lane = wave lane; holds column ct*16 + (lane&15), K rows kk*32 + (lane>>4)*8 + (t&7) + (t>>3)*16
// => each lane's per-step fragment is 16 contiguous halfs (32 B).
__global__ void pack_b_kernel(const float* __restrict__ Wsrc,
                              const float* __restrict__ Wdst,
                              _Float16* __restrict__ Bpack) {
  int i = blockIdx.x * blockDim.x + threadIdx.x;   // 0 .. 256*128-1
  if (i >= 256 * 128) return;
  int t    = i & 15;
  int lane = (i >> 4) & 31;
  int ct   = (i >> 9) & 7;
  int kk   = i >> 12;
  int k    = kk * 32 + (lane >> 4) * 8 + (t & 7) + (t >> 3) * 16;
  int c    = ct * 16 + (lane & 15);
  float v  = (c < 64) ? Wsrc[k * 64 + c] : Wdst[k * 64 + (c - 64)];
  Bpack[i] = (_Float16)v;
}

// ------------------------------------------------------- WMMA GEMM (fused ELU)
// Y[N,128] = elu(X[N,256]) @ [W_src|W_dst], f32 accumulate.
// Block: 256 threads = 8 waves; block covers 16 rows, wave w covers cols [16w,16w+16).
// Whole 16x256 A row-block staged once in LDS (8 KB), then 8 unrolled WMMAs.
__global__ void __launch_bounds__(256)
gemm_elu_proj_kernel(const float* __restrict__ X,
                     const _Float16* __restrict__ Bpack,
                     float* __restrict__ Y, int N) {
  __shared__ _Float16 sA[16 * 256];   // 8 KB

  const int row0 = blockIdx.x * 16;
  const int wave = threadIdx.x >> 5;
  const int lane = threadIdx.x & 31;
  const int col0 = wave * 16;

  // ---- stage full A row-block with fused ELU; 4096 halfs, v2h packed stores
#pragma unroll
  for (int it = 0; it < 8; ++it) {
    int e = (threadIdx.x + it * 256) * 2;        // even half index
    int r = e >> 8, c = e & 255;
    int rr = row0 + r;
    float2 x2 = make_float2(0.f, 0.f);
    if (rr < N) x2 = *(const float2*)(X + (size_t)rr * 256 + c);
    v2h p;
    p[0] = (_Float16)elu_f(x2.x);
    p[1] = (_Float16)elu_f(x2.y);
    *(v2h*)(&sA[e]) = p;
  }
  __syncthreads();

  // ---- fragment bases (ISA 16-bit layouts, wave32)
  const int mA  = lane & 15;          // A row
  const int kbA = (lane >> 4) * 8;    // A K-base (lanes 16-31 -> +8)
  const int nB  = lane & 15;          // B column within tile
  const _Float16* __restrict__ bp = Bpack + ((size_t)wave * 32 + lane) * 16;

  v8f acc0 = {}, acc1 = {};

#pragma unroll
  for (int kk = 0; kk < 8; ++kk) {
    // A 16x32 f16 fragment: two contiguous 16 B LDS chunks per lane
    const _Float16* ap = &sA[mA * 256 + kk * 32 + kbA];
    v8h alo = *(const v8h*)ap;          // K = kb .. kb+7
    v8h ahi = *(const v8h*)(ap + 16);   // K = kb+16 .. kb+23
    v16h a = __builtin_shufflevector(alo, ahi,
        0, 1, 2, 3, 4, 5, 6, 7, 8, 9, 10, 11, 12, 13, 14, 15);
    // B 32x16 f16 fragment: 32 contiguous bytes per lane (pre-packed)
    v16h b = *(const v16h*)(bp + (size_t)kk * (8 * 32 * 16));

    if (kk & 1)
      acc1 = __builtin_amdgcn_wmma_f32_16x16x32_f16(false, a, false, b,
                                                    (short)0, acc1, false, false);
    else
      acc0 = __builtin_amdgcn_wmma_f32_16x16x32_f16(false, a, false, b,
                                                    (short)0, acc0, false, false);
  }

  // ---- epilogue: C/D layout VGPR r -> M = r + 8*(lane>=16), N = lane&15
  const int mb = (lane >> 4) * 8;
#pragma unroll
  for (int r = 0; r < 8; ++r) {
    int rr = row0 + mb + r;
    if (rr < N) Y[(size_t)rr * 128 + col0 + nB] = acc0[r] + acc1[r];
  }
}

// ------------------------------------------- attention coefficients per (n,h)
__global__ void attn_coef_kernel(const float* __restrict__ xsxd,
                                 const float* __restrict__ att_src,
                                 const float* __restrict__ att_dst,
                                 float* __restrict__ a_s,
                                 float* __restrict__ a_d, int N) {
  int t = blockIdx.x * blockDim.x + threadIdx.x;
  if (t >= N * 2) return;
  int n = t >> 1, h = t & 1;
  const float* xr = xsxd + (size_t)n * 128;
  float ss = 0.f, sd = 0.f;
#pragma unroll 8
  for (int c = 0; c < 32; ++c) {
    ss += xr[h * 32 + c]       * att_src[h * 32 + c];
    sd += xr[64 + h * 32 + c]  * att_dst[h * 32 + c];
  }
  a_s[t] = ss;
  a_d[t] = sd;
}

// -------------------------------------------------------------- segment max
__global__ void edge_max_kernel(const int* __restrict__ srcIdx,
                                const int* __restrict__ dstIdx, int E, int N,
                                const float* __restrict__ a_s,
                                const float* __restrict__ a_d,
                                unsigned* __restrict__ m_ord) {
  int e = blockIdx.x * blockDim.x + threadIdx.x;
  if (e >= E + N) return;
  int s = (e < E) ? srcIdx[e] : (e - E);   // self-loops appended
  int d = (e < E) ? dstIdx[e] : (e - E);
#pragma unroll
  for (int h = 0; h < 2; ++h) {
    float ev = lrelu_f(a_s[s * 2 + h] + a_d[d * 2 + h]);
    atomicMax(&m_ord[d * 2 + h], f2ord(ev));
  }
}

// -------------------------------------------------------------- segment sum
__global__ void edge_sum_kernel(const int* __restrict__ srcIdx,
                                const int* __restrict__ dstIdx, int E, int N,
                                const float* __restrict__ a_s,
                                const float* __restrict__ a_d,
                                const unsigned* __restrict__ m_ord,
                                float* __restrict__ ssum) {
  int e = blockIdx.x * blockDim.x + threadIdx.x;
  if (e >= E + N) return;
  int s = (e < E) ? srcIdx[e] : (e - E);
  int d = (e < E) ? dstIdx[e] : (e - E);
#pragma unroll
  for (int h = 0; h < 2; ++h) {
    float ev = lrelu_f(a_s[s * 2 + h] + a_d[d * 2 + h]);
    float p  = expf(ev - ord2f(m_ord[d * 2 + h]));
    atomicAdd(&ssum[d * 2 + h], p);
  }
}

// --------------------------------------------- scatter alpha * xs[src] -> out
// 64 threads per edge (one per output channel), 4 edges per 256-thread block.
__global__ void __launch_bounds__(256)
aggregate_kernel(const int* __restrict__ srcIdx, const int* __restrict__ dstIdx,
                 int E, int N,
                 const float* __restrict__ a_s, const float* __restrict__ a_d,
                 const unsigned* __restrict__ m_ord,
                 const float* __restrict__ ssum,
                 const float* __restrict__ xsxd_src,
                 float* __restrict__ out) {
  int e = blockIdx.x * 4 + (threadIdx.x >> 6);
  if (e >= E + N) return;
  int cc = threadIdx.x & 63;
  int h  = cc >> 5;
  int s = (e < E) ? srcIdx[e] : (e - E);
  int d = (e < E) ? dstIdx[e] : (e - E);
  float ev    = lrelu_f(a_s[s * 2 + h] + a_d[d * 2 + h]);
  float alpha = expf(ev - ord2f(m_ord[d * 2 + h])) / (ssum[d * 2 + h] + 1e-16f);
  atomicAdd(&out[(size_t)d * 64 + cc], alpha * xsxd_src[(size_t)s * 128 + cc]);
}

// --------------------------------------------------- enrichment mean + epilogue
__global__ void reduce_sum_kernel(const float* __restrict__ w, int n,
                                  float* __restrict__ red) {
  __shared__ float sm[256];
  float acc = 0.f;
  for (int i = blockIdx.x * blockDim.x + threadIdx.x; i < n;
       i += gridDim.x * blockDim.x)
    acc += w[i];
  sm[threadIdx.x] = acc;
  __syncthreads();
  for (int s = 128; s > 0; s >>= 1) {
    if (threadIdx.x < s) sm[threadIdx.x] += sm[threadIdx.x + s];
    __syncthreads();
  }
  if (threadIdx.x == 0) atomicAdd(&red[0], sm[0]);
}

__global__ void compute_eb_kernel(const float* __restrict__ red,
                                  const float* __restrict__ escale,
                                  float Einv, float* __restrict__ eb) {
  float mean = red[0] * Einv;
  float avg  = fminf(fmaxf(mean, 0.2f), 5.0f);
  float sf   = 0.1f / (1.f + expf(-escale[0]));
  eb[0] = sf * (avg - 1.f);
}

__global__ void finalize_kernel(float* __restrict__ out,
                                const float* __restrict__ bias,
                                const float* __restrict__ eb, int n) {
  int i = blockIdx.x * blockDim.x + threadIdx.x;
  if (i < n) out[i] += bias[i & 63] + eb[0];
}

// ---------------------------------------------------------------------------
extern "C" void kernel_launch(void* const* d_in, const int* in_sizes, int n_in,
                              void* d_out, int out_size, void* d_ws, size_t ws_size,
                              hipStream_t stream) {
  (void)n_in; (void)ws_size;
  const float* h_x     = (const float*)d_in[0];
  const float* t_x     = (const float*)d_in[1];
  const float* W_src   = (const float*)d_in[2];
  const float* W_dst   = (const float*)d_in[3];
  const float* att_src = (const float*)d_in[4];
  const float* att_dst = (const float*)d_in[5];
  const float* bias    = (const float*)d_in[6];
  const float* ew      = (const float*)d_in[7];
  const float* escale  = (const float*)d_in[8];
  const int*   edges   = (const int*)d_in[9];

  const int D = 256;
  const int N = in_sizes[0] / D;       // 50000
  const int E = in_sizes[7];           // 800000
  const int* row = edges;              // edge_index[0]
  const int* col = edges + E;          // edge_index[1]

  // ---- workspace carve-out (256B aligned) ----
  char* wp = (char*)d_ws;
  auto carve = [&](size_t bytes) -> char* {
    char* p = wp;
    wp += (bytes + 255) & ~(size_t)255;
    return p;
  };
  float*     xsxd_h = (float*)carve((size_t)N * 128 * 4);   // elu(h)@[Ws|Wd]
  float*     xsxd_t = (float*)carve((size_t)N * 128 * 4);   // elu(t)@[Ws|Wd]
  _Float16*  Bpack  = (_Float16*)carve((size_t)256 * 128 * 2);
  float*     a_sh   = (float*)carve((size_t)N * 2 * 4);
  float*     a_dh   = (float*)carve((size_t)N * 2 * 4);
  float*     a_st   = (float*)carve((size_t)N * 2 * 4);
  float*     a_dt   = (float*)carve((size_t)N * 2 * 4);
  unsigned*  m_t    = (unsigned*)carve((size_t)N * 2 * 4);
  unsigned*  m_h    = (unsigned*)carve((size_t)N * 2 * 4);
  float*     s_t    = (float*)carve((size_t)N * 2 * 4);
  float*     s_h    = (float*)carve((size_t)N * 2 * 4);
  float*     red    = (float*)carve(8 * 4);   // red[0]=sum(ew), red[1]=eb

  float* out_h = (float*)d_out;                  // h_rep first in tuple
  float* out_t = out_h + (size_t)N * 64;         // then t_rep
  (void)out_size;

  const int B = 256;
  auto g = [](int n, int b) { return (n + b - 1) / b; };

  // 1) weights -> f16, pre-swizzled to WMMA-B fragment order
  pack_b_kernel<<<g(256 * 128, B), B, 0, stream>>>(W_src, W_dst, Bpack);

  // 2) init accumulators
  fill_f32_kernel<<<g(2 * N * 64, B), B, 0, stream>>>((float*)d_out, 0.f, 2 * N * 64);
  fill_f32_kernel<<<g(N * 2, B), B, 0, stream>>>(s_t, 0.f, N * 2);
  fill_f32_kernel<<<g(N * 2, B), B, 0, stream>>>(s_h, 0.f, N * 2);
  fill_ord_kernel<<<g(N * 2, B), B, 0, stream>>>(m_t, -3.0e38f, N * 2);
  fill_ord_kernel<<<g(N * 2, B), B, 0, stream>>>(m_h, -3.0e38f, N * 2);
  fill_f32_kernel<<<1, 8, 0, stream>>>(red, 0.f, 8);

  // 3) WMMA projections
  gemm_elu_proj_kernel<<<g(N, 16), B, 0, stream>>>(h_x, Bpack, xsxd_h, N);
  gemm_elu_proj_kernel<<<g(N, 16), B, 0, stream>>>(t_x, Bpack, xsxd_t, N);

  // 4) attention coefficients
  attn_coef_kernel<<<g(N * 2, B), B, 0, stream>>>(xsxd_h, att_src, att_dst, a_sh, a_dh, N);
  attn_coef_kernel<<<g(N * 2, B), B, 0, stream>>>(xsxd_t, att_src, att_dst, a_st, a_dt, N);

  const int Ep = E + N;  // with self loops

  // 5) t_rep direction: src = h (row), dst = t (col)
  edge_max_kernel<<<g(Ep, B), B, 0, stream>>>(row, col, E, N, a_sh, a_dt, m_t);
  edge_sum_kernel<<<g(Ep, B), B, 0, stream>>>(row, col, E, N, a_sh, a_dt, m_t, s_t);
  aggregate_kernel<<<g(Ep, 4), B, 0, stream>>>(row, col, E, N, a_sh, a_dt, m_t, s_t,
                                               xsxd_h, out_t);

  // 6) h_rep direction: src = t (col), dst = h (row)
  edge_max_kernel<<<g(Ep, B), B, 0, stream>>>(col, row, E, N, a_st, a_dh, m_h);
  edge_sum_kernel<<<g(Ep, B), B, 0, stream>>>(col, row, E, N, a_st, a_dh, m_h, s_h);
  aggregate_kernel<<<g(Ep, 4), B, 0, stream>>>(col, row, E, N, a_st, a_dh, m_h, s_h,
                                               xsxd_t, out_h);

  // 7) enrichment bias + epilogue
  reduce_sum_kernel<<<512, B, 0, stream>>>(ew, E, red);
  compute_eb_kernel<<<1, 1, 0, stream>>>(red, escale, 1.0f / (float)E, red + 1);
  finalize_kernel<<<g(2 * N * 64, B), B, 0, stream>>>((float*)d_out, bias, red + 1,
                                                      2 * N * 64);
}
